// LlamaMoC_mixed_6579889898129
// MI455X (gfx1250) — matrix operand order repaired and checked by reading, not compiled, not run on hardware
//
#include <hip/hip_runtime.h>
#include <hip/hip_bf16.h>

typedef __attribute__((ext_vector_type(16))) __bf16 v16bf;
typedef __attribute__((ext_vector_type(8)))  __bf16 v8bf;
typedef __attribute__((ext_vector_type(8)))  float  v8f;

// ---------------------------------------------------------------------------
// CDNA5 async global->LDS (no VGPR round-trip, tracked by ASYNCcnt).
// lds_off = LDS byte offset (low 32 bits of generic pointer to __shared__).
// ---------------------------------------------------------------------------
__device__ __forceinline__ void async_load_b128(unsigned lds_off, const void* gaddr) {
  asm volatile("global_load_async_to_lds_b128 %0, %1, off"
               :: "v"(lds_off), "v"(gaddr)
               : "memory");
}
__device__ __forceinline__ void wait_asynccnt0() {
  asm volatile("s_wait_asynccnt 0x0" ::: "memory");
}

// ---------------------------------------------------------------------------
// fp32 -> bf16 conversion (bandwidth-bound, grid-stride)
// ---------------------------------------------------------------------------
__global__ __launch_bounds__(256) void cvt_f32_bf16(const float* __restrict__ s,
                                                    __bf16* __restrict__ d, int n) {
  for (int i = blockIdx.x * blockDim.x + threadIdx.x; i < n;
       i += gridDim.x * blockDim.x)
    d[i] = (__bf16)s[i];
}

// ---------------------------------------------------------------------------
// C[M,N] = A[M,K] * Bw[N,K]^T   (A,Bw bf16 row-major; C f32 or bf16)
// 128x128 tile / block (256 threads = 8 waves). Double-buffered KT=64 LDS
// stages filled with global_load_async_to_lds_b128; each wave computes a
// 32x64 sub-tile = 2x4 tiles of 16x16 via v_wmma_f32_16x16x32_bf16.
// M,N multiples of 128; K multiple of 64.
// ---------------------------------------------------------------------------
template <bool OUT_BF16>
__global__ __launch_bounds__(256) void gemm_bf16(const __bf16* __restrict__ A,
                                                 const __bf16* __restrict__ Bw,
                                                 void* __restrict__ Cv,
                                                 int M, int N, int K) {
  constexpr int KT   = 64;
  constexpr int LSTR = KT + 8;  // 72 elems = 144 B row stride (16B aligned)
  __shared__ __align__(16) __bf16 As[2][128 * LSTR];
  __shared__ __align__(16) __bf16 Bs[2][128 * LSTR];

  const int tid  = threadIdx.x;
  const int lane = tid & 31;
  const int wid  = tid >> 5;
  const int lm   = lane & 15;   // M row (A frag) / N col (B frag) / N col (C)
  const int lh   = lane >> 4;   // which K/M half the lane holds
  const int waveM  = (wid & 3) * 32;
  const int waveN  = (wid >> 2) * 64;
  const int blockN = blockIdx.x * 128;
  const int blockM = blockIdx.y * 128;

  v8f acc[2][4];
  for (int r = 0; r < 2; ++r)
    for (int c = 0; c < 4; ++c)
      for (int i = 0; i < 8; ++i) acc[r][c][i] = 0.f;

  // issue all async loads for one KT-tile into LDS buffer `buf`
  auto issue_tile = [&](int buf, int k0) {
    for (int ch = tid; ch < 1024; ch += 256) {      // A: 128 rows x 8 chunks
      const int row = ch >> 3, kc = (ch & 7) * 8;
      async_load_b128((unsigned)(size_t)(void*)&As[buf][row * LSTR + kc],
                      (const void*)(A + (size_t)(blockM + row) * K + k0 + kc));
    }
    for (int ch = tid; ch < 1024; ch += 256) {      // B: 128 rows x 8 chunks
      const int row = ch >> 3, kc = (ch & 7) * 8;
      async_load_b128((unsigned)(size_t)(void*)&Bs[buf][row * LSTR + kc],
                      (const void*)(Bw + (size_t)(blockN + row) * K + k0 + kc));
    }
  };

  const int ntiles = K / KT;
  issue_tile(0, 0);

  for (int t = 0; t < ntiles; ++t) {
    const int cur = t & 1;
    wait_asynccnt0();    // my loads for tile t (and earlier) are in LDS
    __syncthreads();     // everyone's tile-t data visible; everyone done with t-1
    if (t + 1 < ntiles)  // prefetch next tile into the other buffer
      issue_tile(cur ^ 1, (t + 1) * KT);

    // ---- two 16x16x32 k-steps per LDS stage ----
    for (int kk = 0; kk < KT; kk += 32) {
      v16bf a[2], b[4];
      // A 16x32 frag: lane(0-15)=row M, holds K {lh*8..+7} and {16+lh*8..+7}
      for (int r = 0; r < 2; ++r) {
        const int row = waveM + r * 16 + lm;
        union { v16bf v; v8bf h[2]; } u;
        u.h[0] = *(const v8bf*)&As[cur][row * LSTR + kk + lh * 8];
        u.h[1] = *(const v8bf*)&As[cur][row * LSTR + kk + 16 + lh * 8];
        a[r] = u.v;
      }
      // B 32x16 frag: lane(0-15)=col N, holds K {lh*8..+7} and {16+lh*8..+7}
      for (int c = 0; c < 4; ++c) {
        const int row = waveN + c * 16 + lm;  // Bw row = output col
        union { v16bf v; v8bf h[2]; } u;
        u.h[0] = *(const v8bf*)&Bs[cur][row * LSTR + kk + lh * 8];
        u.h[1] = *(const v8bf*)&Bs[cur][row * LSTR + kk + 16 + lh * 8];
        b[c] = u.v;
      }
      for (int r = 0; r < 2; ++r)
        for (int c = 0; c < 4; ++c)
          acc[r][c] = __builtin_amdgcn_wmma_f32_16x16x32_bf16(
              false, a[r], false, b[c], (short)0, acc[r][c], false, false);
    }
    __syncthreads();     // done reading buf `cur` before it is refilled at t+2
  }

  // ---- store: VGPR i -> (M = base + lh*8 + i, N = lm) ----
  for (int r = 0; r < 2; ++r)
    for (int c = 0; c < 4; ++c) {
      const int row0 = blockM + waveM + r * 16 + lh * 8;
      const int col  = blockN + waveN + c * 16 + lm;
      if (OUT_BF16) {
        __bf16* Cb = (__bf16*)Cv;
        for (int i = 0; i < 8; ++i)
          Cb[(size_t)(row0 + i) * N + col] = (__bf16)acc[r][c][i];
      } else {
        float* Cf = (float*)Cv;
        for (int i = 0; i < 8; ++i)
          Cf[(size_t)(row0 + i) * N + col] = acc[r][c][i];
      }
    }
}

// ---------------------------------------------------------------------------
// Per-token top-K threshold (4-pass 256-bucket radix select on order-preserving
// uint keys cached in LDS) + silu(gate)*v masked write. One block per token.
// Reads the 32KB gate row from HBM exactly once.
// ---------------------------------------------------------------------------
__device__ __forceinline__ unsigned orderKey(float f) {
  unsigned u = __float_as_uint(f);
  return (u & 0x80000000u) ? ~u : (u | 0x80000000u);
}
__device__ __forceinline__ float keyToFloat(unsigned key) {
  unsigned u = (key & 0x80000000u) ? (key & 0x7FFFFFFFu) : ~key;
  return __uint_as_float(u);
}

#define MAX_I 8192

__global__ __launch_bounds__(256) void topk_silu(const float* __restrict__ gate,
                                                 const __bf16* __restrict__ v,
                                                 __bf16* __restrict__ scaled,
                                                 const int* __restrict__ actK,
                                                 int I) {
  const int n   = blockIdx.x;
  const int tid = threadIdx.x;
  const float* grow = gate + (size_t)n * I;

  __shared__ unsigned keys[MAX_I];
  __shared__ unsigned hist[256];
  __shared__ unsigned sPref;
  __shared__ int      sRem;

  for (int i = tid; i < I; i += 256) keys[i] = orderKey(grow[i]);

  unsigned prefix = 0;
  int remain = actK[0];

  for (int pass = 0; pass < 4; ++pass) {
    const int shift = 24 - pass * 8;
    hist[tid] = 0;
    __syncthreads();
    const unsigned mask = (pass == 0) ? 0u : (0xFFFFFFFFu << (shift + 8));
    for (int i = tid; i < I; i += 256) {
      const unsigned key = keys[i];
      if ((key & mask) == prefix) atomicAdd(&hist[(key >> shift) & 255], 1u);
    }
    __syncthreads();
    if (tid == 0) {
      int acc = 0, b;
      for (b = 255; b > 0; --b) {
        const int c = (int)hist[b];
        if (acc + c >= remain) break;
        acc += c;
      }
      sPref = prefix | ((unsigned)b << shift);
      sRem  = remain - acc;
    }
    __syncthreads();
    prefix = sPref;
    remain = sRem;
    __syncthreads();
  }

  const unsigned thresh = prefix;  // exact 32-bit key of K-th largest
  const __bf16* vrow = v + (size_t)n * I;
  __bf16* srow = scaled + (size_t)n * I;
  for (int i = tid; i < I; i += 256) {
    const unsigned key = keys[i];
    float o = 0.f;
    if (key >= thresh) {
      const float g  = keyToFloat(key);
      const float vv = (float)vrow[i];
      o = g * (1.f / (1.f + __expf(-g))) * vv;  // silu(g)*v
    }
    srow[i] = (__bf16)o;
  }
}

// ---------------------------------------------------------------------------
extern "C" void kernel_launch(void* const* d_in, const int* in_sizes, int n_in,
                              void* d_out, int out_size, void* d_ws, size_t ws_size,
                              hipStream_t stream) {
  const float* x  = (const float*)d_in[0];
  const float* Wg = (const float*)d_in[1];
  const float* Wu = (const float*)d_in[2];
  const float* Wd = (const float*)d_in[3];
  const int* actK = (const int*)d_in[4];

  const int H    = 2048;
  const int I    = in_sizes[1] / H;  // 8192
  const int Ntok = in_sizes[0] / H;  // 8192

  const size_t nXH = (size_t)Ntok * H;
  const size_t nIH = (size_t)I * H;
  const size_t nNI = (size_t)Ntok * I;

  char* w = (char*)d_ws;
  __bf16* xb   = (__bf16*)w; w += nXH * 2;
  __bf16* Wgb  = (__bf16*)w; w += nIH * 2;
  __bf16* Wub  = (__bf16*)w; w += nIH * 2;
  __bf16* Wdb  = (__bf16*)w; w += nIH * 2;
  float*  gate = (float*) w; w += nNI * 4;
  __bf16* vb   = (__bf16*)w; w += nNI * 2;
  __bf16* sc   = (__bf16*)w;

  cvt_f32_bf16<<<2048, 256, 0, stream>>>(x,  xb,  (int)nXH);
  cvt_f32_bf16<<<2048, 256, 0, stream>>>(Wg, Wgb, (int)nIH);
  cvt_f32_bf16<<<2048, 256, 0, stream>>>(Wu, Wub, (int)nIH);
  cvt_f32_bf16<<<2048, 256, 0, stream>>>(Wd, Wdb, (int)nIH);

  dim3 gUp(I / 128, Ntok / 128);
  gemm_bf16<false><<<gUp, 256, 0, stream>>>(xb, Wgb, gate, Ntok, I, H);
  gemm_bf16<true ><<<gUp, 256, 0, stream>>>(xb, Wub, vb,   Ntok, I, H);

  topk_silu<<<Ntok, 256, 0, stream>>>(gate, vb, sc, actK, I);

  dim3 gDn(H / 128, Ntok / 128);
  gemm_bf16<false><<<gDn, 256, 0, stream>>>(sc, Wdb, (float*)d_out, Ntok, H, I);
}
